// SparseJanossy_62122406969953
// MI455X (gfx1250) — matrix operand order repaired and checked by reading, not compiled, not run on hardware
//
#include <hip/hip_runtime.h>

#define N_NODES 100000
#define F       128
#define E_EDGES 1600000
#define KARY    8
#define NPERMS  4
#define NOUT    64
#define G4      (4*F)            // 512 gate dim
#define TILES   (N_NODES/16)     // 6250
#define WAVES_PER_BLOCK 8
#define BLOCK   (32*WAVES_PER_BLOCK)

typedef __attribute__((ext_vector_type(16))) _Float16 v16h;
typedef __attribute__((ext_vector_type(8)))  float    v8f;

union FragH { v16h v; uint4 q[2]; _Float16 h[16]; };

__device__ __forceinline__ float sigmoidf_(float x) {
  return __builtin_amdgcn_rcpf(1.0f + __expf(-x));          // v_exp + v_rcp
}
__device__ __forceinline__ float tanhf_(float x) {
  float e = __expf(2.0f * x);
  return 1.0f - 2.0f * __builtin_amdgcn_rcpf(e + 1.0f);      // (e-1)/(e+1)
}
__device__ __forceinline__ v8f splat8(float x) { v8f v = {x,x,x,x,x,x,x,x}; return v; }
__device__ __forceinline__ unsigned short f2h(float x) {
  _Float16 h = (_Float16)x;
  return __builtin_bit_cast(unsigned short, h);
}

// ---------------------------------------------------------------------------
// Kernel 1: neighbor sampling (capped atomic append over hash-permuted edges)
// ---------------------------------------------------------------------------
__global__ void k_build_nbrs(const int* __restrict__ adj,
                             int* __restrict__ nbr, int* __restrict__ cnt) {
  long long gid = (long long)blockIdx.x * blockDim.x + threadIdx.x;
  long long total = (long long)NPERMS * E_EDGES;
  if (gid >= total) return;
  int p = (int)(gid / E_EDGES);
  long long e = gid - (long long)p * E_EDGES;
  const unsigned mults[NPERMS] = {368293u, 1000003u, 999983u, 2718281u}; // coprime with E
  unsigned long long ep = ((unsigned long long)e * mults[p]) % (unsigned long long)E_EDGES;
  int src = adj[2*ep + 0];
  int dst = adj[2*ep + 1];
  if ((unsigned)src >= (unsigned)N_NODES) return;
  int slot = atomicAdd(&cnt[p*N_NODES + src], 1);
  if (slot < KARY) nbr[((long long)(p*N_NODES + src))*KARY + slot] = dst;
}

// ---------------------------------------------------------------------------
// Kernel 2: pack weights into f16 WMMA B-fragment order; bias sum.
// LSTM weights laid out [kt][nt][lane][16] so in-kernel DS offsets are
// base(kt) + nt*1024B immediates (fits 16-bit DS offset).
// ---------------------------------------------------------------------------
__global__ void k_pack(const float* __restrict__ wih, const float* __restrict__ whh,
                       const float* __restrict__ bih, const float* __restrict__ bhh,
                       const float* __restrict__ wout,
                       unsigned short* __restrict__ wihF, unsigned short* __restrict__ whhF,
                       unsigned short* __restrict__ woutF, float* __restrict__ bsum) {
  int gid = blockIdx.x * blockDim.x + threadIdx.x;
  if (gid < 65536) {             // 4 kt * 32 nt * 32 lanes * 16 halves
    int hv   = gid & 15;
    int lane = (gid >> 4) & 31;
    int ntk  = gid >> 9;         // kt*32 + nt
    int nt   = ntk & 31;
    int kt   = ntk >> 5;
    int col  = nt*16 + (lane & 15);                    // gate row
    int k    = kt*32 + ((lane >> 4) & 1)*8 + (hv & 7) + (hv >> 3)*16;      // contraction index
    wihF[gid] = f2h(wih[col*F + k]);
    whhF[gid] = f2h(whh[col*F + k]);
  }
  if (gid < 16384) {             // head: [nt][kt][lane][16]
    int hv   = gid & 15;
    int lane = (gid >> 4) & 31;
    int kt   = (gid >> 9) & 7;
    int nt   = gid >> 12;
    int n    = nt*16 + (lane & 15);
    int k    = kt*32 + ((lane >> 4) & 1)*8 + (hv & 7) + (hv >> 3)*16;
    woutF[gid] = f2h(wout[k*NOUT + n]);
  }
  if (gid < G4) bsum[gid] = bih[gid] + bhh[gid];
}

// ---------------------------------------------------------------------------
// Main kernel: 1 wave = one 16-node tile; LSTM via v_wmma_f32_16x16x32_f16.
// 8 waves/block (2 per SIMD32) share 256KB of LDS weight fragments.
// ---------------------------------------------------------------------------
__launch_bounds__(BLOCK, 2)
__global__ void k_janossy(const float* __restrict__ node_feat,
                          const int* __restrict__ nbr, const int* __restrict__ cnt,
                          const unsigned short* __restrict__ wihF,
                          const unsigned short* __restrict__ whhF,
                          const unsigned short* __restrict__ woutF,
                          const float* __restrict__ bsum,
                          const float* __restrict__ bias,
                          float* __restrict__ out) {
  extern __shared__ unsigned short wdyn[];          // 256 KB: wih + whh fragments
  unsigned short* wih_s = wdyn;
  unsigned short* whh_s = wdyn + 65536;
  __shared__ __align__(32) unsigned short h_lds[WAVES_PER_BLOCK][16*F];  // 32 KB

  const int tid  = threadIdx.x;
  const int wave = tid >> 5;
  const int lane = tid & 31;
  const int l15  = lane & 15;
  const int hi   = (lane >> 4) & 1;

  // stage f16 weight fragments into LDS (8192 uint4 each), reused for all work
  {
    const uint4* sa = (const uint4*)wihF;
    const uint4* sb = (const uint4*)whhF;
    uint4* da = (uint4*)wih_s;
    uint4* db = (uint4*)whh_s;
    for (int i = tid; i < 8192; i += BLOCK) { da[i] = sa[i]; db[i] = sb[i]; }
  }
  __syncthreads();   // only block-wide barrier: weights are read-only afterwards

  int tile = blockIdx.x * WAVES_PER_BLOCK + wave;
  bool valid = (tile < TILES);
  int base = (valid ? tile : (TILES - 1)) * 16;

  // scalar biases kept in registers (value depends only on lane%16 and nt)
  float bs[32];
#pragma unroll
  for (int i = 0; i < 32; ++i) bs[i] = bsum[i*16 + l15];

  v8f c_t[8], hn_t[8];
#pragma unroll
  for (int i = 0; i < 8; ++i) hn_t[i] = splat8(0.0f);

  for (int p = 0; p < NPERMS; ++p) {
    // sequence lengths for the 8 node-rows this lane owns in C-layout
    int len8[8];
#pragma unroll
    for (int r = 0; r < 8; ++r)
      len8[r] = min(cnt[p*N_NODES + base + (r + hi*8)], KARY);

    {
      uint4 z; z.x = z.y = z.z = z.w = 0u;
      uint4* hz = (uint4*)&h_lds[wave][0];
      for (int i = lane; i < 256; i += 32) hz[i] = z;          // h = 0
    }
#pragma unroll
    for (int i = 0; i < 8; ++i) c_t[i] = splat8(0.0f);
    __builtin_amdgcn_wave_barrier();

    const long long nbase = ((long long)(p*N_NODES + base + l15))*KARY;

    for (int t = 0; t < KARY; ++t) {
      // gather this step's neighbor row straight into A-fragment registers
      int idx = nbr[nbase + t];
      if ((unsigned)idx >= (unsigned)N_NODES) idx = 0;
      const float* prow = node_feat + (long long)idx * F;

      // prefetch next step's neighbor row while this step's WMMAs run
      if (t + 1 < KARY) {
        int idx2 = nbr[nbase + t + 1];
        if ((unsigned)idx2 >= (unsigned)N_NODES) idx2 = 0;
        __builtin_prefetch(node_feat + (long long)idx2 * F + hi*64, 0, 1);
      }

      FragH ax[4], ah[4];
#pragma unroll
      for (int kt = 0; kt < 4; ++kt) {
        int c0 = kt*32 + hi*8;
        float4 a0 = *(const float4*)(prow + c0);
        float4 a1 = *(const float4*)(prow + c0 + 4);
        float4 b0 = *(const float4*)(prow + c0 + 16);
        float4 b1 = *(const float4*)(prow + c0 + 20);
        FragH u;
        u.h[ 0]=(_Float16)a0.x; u.h[ 1]=(_Float16)a0.y; u.h[ 2]=(_Float16)a0.z; u.h[ 3]=(_Float16)a0.w;
        u.h[ 4]=(_Float16)a1.x; u.h[ 5]=(_Float16)a1.y; u.h[ 6]=(_Float16)a1.z; u.h[ 7]=(_Float16)a1.w;
        u.h[ 8]=(_Float16)b0.x; u.h[ 9]=(_Float16)b0.y; u.h[10]=(_Float16)b0.z; u.h[11]=(_Float16)b0.w;
        u.h[12]=(_Float16)b1.x; u.h[13]=(_Float16)b1.y; u.h[14]=(_Float16)b1.z; u.h[15]=(_Float16)b1.w;
        ax[kt] = u;
        const unsigned short* ph = &h_lds[wave][l15*F + c0];
        ah[kt].q[0] = *(const uint4*)ph;
        ah[kt].q[1] = *(const uint4*)(ph + 16);
      }
      __builtin_amdgcn_wave_barrier();   // ah reads before h_lds rewrites below

      // gates = x@Wih^T + h@Whh^T : 256 WMMAs per step per wave (C starts at 0)
#pragma unroll
      for (int ft = 0; ft < 8; ++ft) {
        v8f acc[4];
#pragma unroll
        for (int g = 0; g < 4; ++g) acc[g] = splat8(0.0f);     // inline-0 SRC2
#pragma unroll
        for (int kt = 0; kt < 4; ++kt) {
          const v16h* bi_base = (const v16h*)wih_s + (kt*32)*32 + lane;
          const v16h* bh_base = (const v16h*)whh_s + (kt*32)*32 + lane;
#pragma unroll
          for (int g = 0; g < 4; ++g) {
            int nt = ft + g*8;
            v16h bi = bi_base[nt*32];  // base(kt) + nt*1024B immediate
            acc[g] = __builtin_amdgcn_wmma_f32_16x16x32_f16(false, ax[kt].v, false, bi,
                                                            (short)0, acc[g], false, false);
            v16h bh = bh_base[nt*32];
            acc[g] = __builtin_amdgcn_wmma_f32_16x16x32_f16(false, ah[kt].v, false, bh,
                                                            (short)0, acc[g], false, false);
          }
        }
        // elementwise LSTM cell update; bias folded in; state frozen at t>=len
#pragma unroll
        for (int r = 0; r < 8; ++r) {
          float iv = sigmoidf_(acc[0][r] + bs[ft]);
          float fv = sigmoidf_(acc[1][r] + bs[ft + 8]);
          float gv = tanhf_  (acc[2][r] + bs[ft + 16]);
          float ov = sigmoidf_(acc[3][r] + bs[ft + 24]);
          float cold = c_t[ft][r];
          float cnew = fv*cold + iv*gv;
          bool m = (t < len8[r]);
          c_t[ft][r] = m ? cnew : cold;
          if (m) {
            float hnew = ov*tanhf_(cnew);
            h_lds[wave][(r + hi*8)*F + ft*16 + l15] = f2h(hnew);
          }
        }
      }
      __builtin_amdgcn_wave_barrier();  // h writes before next step's ah reads
    } // t

#pragma unroll
    for (int i = 0; i < 8; ++i) hn_t[i] += c_t[i];
  } // p

  // ---- output head: sigmoid([x, h_neigh/4] @ weight + bias), 32 WMMAs ----
#pragma unroll
  for (int ft = 0; ft < 8; ++ft)
#pragma unroll
    for (int r = 0; r < 8; ++r)
      h_lds[wave][(r + hi*8)*F + ft*16 + l15] = f2h(hn_t[ft][r] * 0.25f);
  __builtin_amdgcn_wave_barrier();

  v8f oacc[4];
#pragma unroll
  for (int nt = 0; nt < 4; ++nt) oacc[nt] = splat8(0.0f);

  const float* prow = node_feat + (long long)(base + l15) * F;
  for (int kt = 0; kt < 8; ++kt) {
    FragH a;
    int c0 = (kt & 3)*32 + hi*8;
    if (kt < 4) {  // own node features, direct from global (L2-resident)
      float4 a0 = *(const float4*)(prow + c0);
      float4 a1 = *(const float4*)(prow + c0 + 4);
      float4 b0 = *(const float4*)(prow + c0 + 16);
      float4 b1 = *(const float4*)(prow + c0 + 20);
      a.h[ 0]=(_Float16)a0.x; a.h[ 1]=(_Float16)a0.y; a.h[ 2]=(_Float16)a0.z; a.h[ 3]=(_Float16)a0.w;
      a.h[ 4]=(_Float16)a1.x; a.h[ 5]=(_Float16)a1.y; a.h[ 6]=(_Float16)a1.z; a.h[ 7]=(_Float16)a1.w;
      a.h[ 8]=(_Float16)b0.x; a.h[ 9]=(_Float16)b0.y; a.h[10]=(_Float16)b0.z; a.h[11]=(_Float16)b0.w;
      a.h[12]=(_Float16)b1.x; a.h[13]=(_Float16)b1.y; a.h[14]=(_Float16)b1.z; a.h[15]=(_Float16)b1.w;
    } else {       // averaged neighbor state from LDS
      const unsigned short* pa = &h_lds[wave][l15*F + c0];
      a.q[0] = *(const uint4*)pa;
      a.q[1] = *(const uint4*)(pa + 16);
    }
#pragma unroll
    for (int nt = 0; nt < 4; ++nt) {
      v16h b = ((const v16h*)woutF)[(nt*8 + kt)*32 + lane];
      oacc[nt] = __builtin_amdgcn_wmma_f32_16x16x32_f16(false, a.v, false, b,
                                                        (short)0, oacc[nt], false, false);
    }
  }

  if (valid) {
#pragma unroll
    for (int nt = 0; nt < 4; ++nt) {
      float bo = bias[nt*16 + l15];
#pragma unroll
      for (int r = 0; r < 8; ++r)
        out[(long long)(base + r + hi*8)*NOUT + nt*16 + l15] = sigmoidf_(oacc[nt][r] + bo);
    }
  }
}

// ---------------------------------------------------------------------------
extern "C" void kernel_launch(void* const* d_in, const int* in_sizes, int n_in,
                              void* d_out, int out_size, void* d_ws, size_t ws_size,
                              hipStream_t stream) {
  (void)in_sizes; (void)n_in; (void)out_size; (void)ws_size;
  const float* node_feat = (const float*)d_in[0];
  const int*   adj       = (const int*)d_in[1];
  const float* w_ih      = (const float*)d_in[2];
  const float* w_hh      = (const float*)d_in[3];
  const float* b_ih      = (const float*)d_in[4];
  const float* b_hh      = (const float*)d_in[5];
  const float* weight    = (const float*)d_in[6];
  const float* bias      = (const float*)d_in[7];
  float* out = (float*)d_out;

  char* ws = (char*)d_ws;
  size_t off = 0;
  int* nbr = (int*)(ws + off);            off += (size_t)NPERMS*N_NODES*KARY*sizeof(int);
  int* cnt = (int*)(ws + off);            off += (size_t)NPERMS*N_NODES*sizeof(int);
  unsigned short* wihF  = (unsigned short*)(ws + off); off += 65536u*2;
  unsigned short* whhF  = (unsigned short*)(ws + off); off += 65536u*2;
  unsigned short* woutF = (unsigned short*)(ws + off); off += 16384u*2;
  float* bsum = (float*)(ws + off);       off += (size_t)G4*sizeof(float);

  hipMemsetAsync(cnt, 0, (size_t)NPERMS*N_NODES*sizeof(int), stream);

  long long totalE = (long long)NPERMS * E_EDGES;
  k_build_nbrs<<<(int)((totalE + 255) / 256), 256, 0, stream>>>(adj, nbr, cnt);

  k_pack<<<(65536 + 255) / 256, 256, 0, stream>>>(w_ih, w_hh, b_ih, b_hh, weight,
                                                  wihF, whhF, woutF, bsum);

  int smem = 2 * 65536 * 2;  // 256 KB dynamic LDS for weight fragments
  hipFuncSetAttribute(reinterpret_cast<const void*>(k_janossy),
                      hipFuncAttributeMaxDynamicSharedMemorySize, smem);
  int blocks = (TILES + WAVES_PER_BLOCK - 1) / WAVES_PER_BLOCK;
  k_janossy<<<blocks, BLOCK, smem, stream>>>(node_feat, nbr, cnt,
                                             wihF, whhF, woutF, bsum, bias, out);
}
